// Attention_67929202754275
// MI455X (gfx1250) — matrix operand order repaired and checked by reading, not compiled, not run on hardware
//
#include <hip/hip_runtime.h>

// Problem constants (match reference)
#define BB  2
#define LL  2048
#define HH  1024
#define NHH 16
#define DD  64
#define KT  32      // k-tile length in flash attention

typedef __bf16 bf16;
typedef __attribute__((ext_vector_type(16))) __bf16 v16bf;
typedef __attribute__((ext_vector_type(8)))  float  v8f;
typedef __attribute__((ext_vector_type(4)))  unsigned int u32x4;
typedef __attribute__((ext_vector_type(8)))  unsigned int u32x8;

union FragBits { v16bf v; unsigned int u[8]; };

// A-fragment (16xK, row-major): lane: row m = lane&15, half h = lane>>4.
// dword v holds bf16 pair at K = kbase + (v&3)*2 + (v>>2)*16 + h*8  (ISA 7.12.2, 16-bit A 16x32)
__device__ __forceinline__ int kmap_a(int v, int h) {
    return ((v & 3) << 1) + ((v >> 2) << 4) + (h << 3);
}
// B-fragment (Kx16, column per lane): lane: col n = lane&15, half h.
// dword v holds bf16 pair at K = kbase + (v&3)*2 + (v>>2)*16 + h*8
__device__ __forceinline__ int kmap_b(int v, int h) {
    return ((v & 3) << 1) + ((v >> 2) << 4) + (h << 3);
}

__device__ __forceinline__ v16bf frag_a_f32(const float* row, int kbase, int h) {
    v16bf f;
#pragma unroll
    for (int v = 0; v < 8; ++v) {
        const float2 p = *(const float2*)(row + kbase + kmap_a(v, h));
        f[2 * v]     = (bf16)p.x;
        f[2 * v + 1] = (bf16)p.y;
    }
    return f;
}
__device__ __forceinline__ v16bf frag_b_f32(const float* row, int kbase, int h) {
    v16bf f;
#pragma unroll
    for (int v = 0; v < 8; ++v) {
        const float2 p = *(const float2*)(row + kbase + kmap_b(v, h));
        f[2 * v]     = (bf16)p.x;
        f[2 * v + 1] = (bf16)p.y;
    }
    return f;
}
__device__ __forceinline__ v16bf frag_a_bf16(const bf16* row, int kbase, int h) {
    FragBits f;
#pragma unroll
    for (int v = 0; v < 8; ++v)
        f.u[v] = *(const unsigned int*)(row + kbase + kmap_a(v, h));
    return f.v;
}
__device__ __forceinline__ v16bf frag_b_bf16(const bf16* row, int kbase, int h) {
    FragBits f;
#pragma unroll
    for (int v = 0; v < 8; ++v)
        f.u[v] = *(const unsigned int*)(row + kbase + kmap_b(v, h));
    return f.v;
}

#define WMMA_BF16(A, Bm, C) \
    __builtin_amdgcn_wmma_f32_16x16x32_bf16(false, (A), false, (Bm), (short)0, (C), false, false)

// ---- CDNA5 async global->LDS copy (ASYNCcnt-tracked, 16B per lane) -------
__device__ __forceinline__ void async_load_b128(unsigned lds_byte_off, const void* gaddr) {
    asm volatile("global_load_async_to_lds_b128 %0, %1, off"
                 :
                 : "v"(lds_byte_off), "v"(gaddr)
                 : "memory");
}
__device__ __forceinline__ void wait_async0() {
    asm volatile("s_wait_asynccnt 0x0" ::: "memory");
}
// Generic-AS pointer to a __shared__ object: low 32 bits are the LDS byte offset
__device__ __forceinline__ unsigned lds_offset_of(const void* p) {
    return (unsigned)(uintptr_t)p;
}

// ---- CDNA5 Tensor Data Mover: one-descriptor 2D tile DMA into LDS --------
// D# group 0 (ISA 8.3): [1:0]=count=1, [63:32]=lds_addr, [120:64]=global_addr,
// [127:126]=type=2.
__device__ __forceinline__ u32x4 tdm_group0(unsigned lds_addr, unsigned long long gaddr) {
    u32x4 g;
    g[0] = 1u;                                          // count=1, user mode
    g[1] = lds_addr;                                    // LDS byte address
    g[2] = (unsigned)(gaddr & 0xFFFFFFFFu);             // global_addr[31:0]
    g[3] = (unsigned)((gaddr >> 32) & 0x1FFFFFFu) | (2u << 30);  // [56:32] | type=2
    return g;
}
// D# group 1 (ISA 8.4): data_size=1 (2B elems), 2D tile (tile_dim2=0),
// dims/strides in element units. No padding, no multicast, no atomic barrier.
__device__ __forceinline__ u32x8 tdm_group1(unsigned tensor_d0, unsigned tensor_d1,
                                            unsigned tile_d0, unsigned tile_d1,
                                            unsigned long long stride0) {
    u32x8 g;
    g[0] = 1u << 16;                                                  // data_size=2B
    g[1] = (tensor_d0 & 0xFFFFu) << 16;                               // tdim0[15:0]@48
    g[2] = ((tensor_d0 >> 16) & 0xFFFFu) | ((tensor_d1 & 0xFFFFu) << 16);
    g[3] = ((tensor_d1 >> 16) & 0xFFFFu) | ((tile_d0 & 0xFFFFu) << 16);
    g[4] = tile_d1 & 0xFFFFu;                                         // tile_dim2=0
    g[5] = (unsigned)(stride0 & 0xFFFFFFFFu);
    g[6] = (unsigned)((stride0 >> 32) & 0xFFFFu);                     // stride1=0
    g[7] = 0u;
    return g;
}
__device__ __forceinline__ void tdm_load_2d(u32x4 g0, u32x8 g1) {
    asm volatile("tensor_load_to_lds %0, %1" :: "s"(g0), "s"(g1) : "memory");
}
__device__ __forceinline__ void wait_tensor0() {
    __builtin_amdgcn_s_wait_tensorcnt(0);   // no-op for waves with TENSORcnt==0
}

// ---------------------------------------------------------------------------
// Kernel 1: fused Q/K/V projections. blockIdx.z selects which projection.
// Q,K stored [B,NH,L,D] bf16; V stored transposed [B,NH,D,L] bf16.
// Q pre-scaled by D^-0.5 = 0.125.
// ---------------------------------------------------------------------------
__global__ void qkv_proj_kernel(const float* __restrict__ x, const float* __restrict__ y,
                                const float* __restrict__ wq, const float* __restrict__ wk,
                                const float* __restrict__ wv,
                                bf16* __restrict__ Qw, bf16* __restrict__ Kw, bf16* __restrict__ Vw) {
    const int lane = threadIdx.x;
    const int col = lane & 15, h = lane >> 4;
    const int which = blockIdx.z;
    const float* in = (which == 0) ? x : y;
    const float* w  = (which == 0) ? wq : ((which == 1) ? wk : wv);
    const int mrow  = blockIdx.x * 16 + col;   // per-lane A row
    const int nbase = blockIdx.y * 64;

    const float* arow = in + (size_t)mrow * HH;

    v8f acc[4];
#pragma unroll
    for (int t = 0; t < 4; ++t)
#pragma unroll
        for (int i = 0; i < 8; ++i) acc[t][i] = 0.0f;

    for (int kb = 0; kb < HH; kb += 32) {
        const v16bf a = frag_a_f32(arow, kb, h);
#pragma unroll
        for (int t = 0; t < 4; ++t) {
            const v16bf bfr = frag_b_f32(w + (size_t)(nbase + t * 16 + col) * HH, kb, h);
            acc[t] = WMMA_BF16(a, bfr, acc[t]);
        }
    }

    // C/D layout: element (m,n): m = tileM*16 + r + 8*h, n = nbase + t*16 + col
#pragma unroll
    for (int t = 0; t < 4; ++t) {
#pragma unroll
        for (int r = 0; r < 8; ++r) {
            const int m = blockIdx.x * 16 + r + 8 * h;
            const int n = nbase + t * 16 + col;
            const int b    = m >> 11;          // / LL
            const int l    = m & (LL - 1);
            const int head = n >> 6;           // / DD
            const int d    = n & (DD - 1);
            const float val = acc[t][r];
            if (which == 0) {
                Qw[(((size_t)b * NHH + head) * LL + l) * DD + d] = (bf16)(val * 0.125f);
            } else if (which == 1) {
                Kw[(((size_t)b * NHH + head) * LL + l) * DD + d] = (bf16)val;
            } else {
                Vw[(((size_t)b * NHH + head) * DD + d) * LL + l] = (bf16)val;   // V^T
            }
        }
    }
}

// ---------------------------------------------------------------------------
// Kernel 2: flash attention. 4 waves per workgroup; each wave owns a 16-row
// q-tile. Per 32-k tile, wave 0 DMAs the K tile (32x64 bf16, one TDM 2D
// descriptor) while all waves async-copy the V^T tile (64x32 bf16) into LDS;
// double-buffered. Each wave reads WMMA fragments from LDS; online softmax;
// P converts C-layout->A-layout via per-wave LDS scratch.
// ---------------------------------------------------------------------------
__global__ __launch_bounds__(128) void
flash_attn_kernel(const bf16* __restrict__ Qw, const bf16* __restrict__ Kw,
                  const bf16* __restrict__ Vt, const float* __restrict__ bias,
                  bf16* __restrict__ Ow) {
    const int tid  = threadIdx.x;         // 0..127
    const int lane = tid & 31;
    const int wave = tid >> 5;            // 0..3
    const int col = lane & 15, h = lane >> 4;
    const int qbase = blockIdx.x * 64 + wave * 16;
    const int head  = blockIdx.y;
    const int b     = blockIdx.z;

    const bf16* Qp = Qw + (((size_t)b * NHH + head) * LL + qbase) * DD;
    const bf16* Kp = Kw + ((size_t)b * NHH + head) * LL * DD;    // [L, D]
    const bf16* Vp = Vt + ((size_t)b * NHH + head) * DD * LL;    // [D, L]

    __shared__ __align__(16) bf16 Klds[2][KT][DD];     // 2 x 4KB, row-major [k][d]
    __shared__ __align__(16) bf16 Vlds[2][DD][KT];     // 2 x 4KB, V^T [d][k]
    __shared__ __align__(16) bf16 Plds[4][16 * KT];    // per-wave P scratch

    // K tile: one TDM 2D descriptor (rows of 64 elems, 32 rows, stride 64 elems).
    auto issue_k_tdm = [&](int buf, int kt) {
        const u32x4 g0 = tdm_group0(lds_offset_of(&Klds[buf][0][0]),
                                    (unsigned long long)(uintptr_t)(Kp + (size_t)kt * DD));
        const u32x8 g1 = tdm_group1(/*tensor_d0=*/DD, /*tensor_d1=*/LL,
                                    /*tile_d0=*/DD,   /*tile_d1=*/KT,
                                    /*stride0=*/DD);
        tdm_load_2d(g0, g1);
    };
    // V^T tile: cooperative async copies (16B per lane, 2 per wave).
    auto issue_v_async = [&](int buf, int kt) {
        const unsigned vbase_lds = lds_offset_of(&Vlds[buf][0][0]);
#pragma unroll
        for (int p = 0; p < 2; ++p) {
            const int d    = (tid >> 2) + 32 * p;                // V^T row
            const int part = tid & 3;                            // 16B chunk in row
            const unsigned loff = vbase_lds + (unsigned)(d * (KT * 2) + part * 16);
            const char* vg = (const char*)(Vp + (size_t)d * LL + kt) + part * 16;
            async_load_b128(loff, vg);
        }
    };

    // Q fragments (d split into two 32-chunks), held in registers for whole loop
    v16bf qf[2];
#pragma unroll
    for (int c = 0; c < 2; ++c) qf[c] = frag_a_bf16(Qp + (size_t)col * DD, 32 * c, h);

    v8f o[4];
#pragma unroll
    for (int t = 0; t < 4; ++t)
#pragma unroll
        for (int i = 0; i < 8; ++i) o[t][i] = 0.0f;
    float mst[8], lst[8];
#pragma unroll
    for (int r = 0; r < 8; ++r) { mst[r] = -1e30f; lst[r] = 0.0f; }

    if (wave == 0) issue_k_tdm(0, 0);
    issue_v_async(0, 0);

    for (int kt = 0, it = 0; kt < LL; kt += KT, ++it) {
        const int buf = it & 1;
        wait_tensor0();         // wave 0's TDM for `buf` landed (no-op for others)
        wait_async0();          // this wave's V async copies for `buf` landed
        __syncthreads();        // publish LDS tile; fences buf^1 readers

        if (kt + KT < LL) {     // prefetch next tile
            if (wave == 0) issue_k_tdm(buf ^ 1, kt + KT);
            issue_v_async(buf ^ 1, kt + KT);
        }

        // S = Q K^T for 16x32 tile (two 16x16 n-tiles, contraction d=64 in 2 chunks)
        v8f s[2];
#pragma unroll
        for (int nt = 0; nt < 2; ++nt)
#pragma unroll
            for (int i = 0; i < 8; ++i) s[nt][i] = 0.0f;
#pragma unroll
        for (int nt = 0; nt < 2; ++nt) {
            const bf16* krow = &Klds[buf][nt * 16 + col][0];
#pragma unroll
            for (int c = 0; c < 2; ++c) {
                const v16bf kf = frag_b_bf16(krow, 32 * c, h);
                s[nt] = WMMA_BF16(qf[c], kf, s[nt]);
            }
        }
        // bias add (reference adds bias[1,1,L,L])
#pragma unroll
        for (int nt = 0; nt < 2; ++nt)
#pragma unroll
            for (int r = 0; r < 8; ++r)
                s[nt][r] += bias[(size_t)(qbase + r + 8 * h) * LL + kt + nt * 16 + col];

        // online softmax: row m = r + 8*h lives across the 16 lanes of a half
        float alpha[8];
#pragma unroll
        for (int r = 0; r < 8; ++r) {
            float mx = fmaxf(s[0][r], s[1][r]);
            mx = fmaxf(mx, __shfl_xor(mx, 1, 32));
            mx = fmaxf(mx, __shfl_xor(mx, 2, 32));
            mx = fmaxf(mx, __shfl_xor(mx, 4, 32));
            mx = fmaxf(mx, __shfl_xor(mx, 8, 32));
            const float mnew = fmaxf(mst[r], mx);
            alpha[r] = expf(mst[r] - mnew);
            mst[r] = mnew;
            const float p0 = expf(s[0][r] - mnew);
            const float p1 = expf(s[1][r] - mnew);
            s[0][r] = p0; s[1][r] = p1;
            float rs = p0 + p1;
            rs += __shfl_xor(rs, 1, 32);
            rs += __shfl_xor(rs, 2, 32);
            rs += __shfl_xor(rs, 4, 32);
            rs += __shfl_xor(rs, 8, 32);
            lst[r] = lst[r] * alpha[r] + rs;
        }
#pragma unroll
        for (int t = 0; t < 4; ++t)
#pragma unroll
            for (int r = 0; r < 8; ++r) o[t][r] *= alpha[r];

        // P: C-layout -> row-major 16x32 bf16 in this wave's LDS scratch
        // (same-wave DS ops are in-order; no barrier needed for the round-trip)
#pragma unroll
        for (int nt = 0; nt < 2; ++nt)
#pragma unroll
            for (int r = 0; r < 8; ++r)
                Plds[wave][(r + 8 * h) * KT + nt * 16 + col] = (bf16)s[nt][r];

        // O += P V  (contraction over 32 k-values from the LDS V^T tile)
        const v16bf pf = frag_a_bf16(&Plds[wave][col * KT], 0, h);
#pragma unroll
        for (int t = 0; t < 4; ++t) {
            const v16bf vf = frag_b_bf16(&Vlds[buf][t * 16 + col][0], 0, h);
            o[t] = WMMA_BF16(pf, vf, o[t]);
        }
        // Next iteration's top barrier guarantees all waves finished reading
        // `buf` before anyone's prefetch overwrites it.
    }

    // normalize rows, store merged heads [B, L, H] as bf16
#pragma unroll
    for (int r = 0; r < 8; ++r) {
        const float inv = 1.0f / lst[r];
        const int l = qbase + r + 8 * h;
#pragma unroll
        for (int t = 0; t < 4; ++t) {
            const int hcol = head * DD + t * 16 + col;
            Ow[((size_t)b * LL + l) * HH + hcol] = (bf16)(o[t][r] * inv);
        }
    }
}

// ---------------------------------------------------------------------------
// Kernel 3: output projection  out = O @ wo^T  (O bf16, wo cvt in-register)
// ---------------------------------------------------------------------------
__global__ void out_proj_kernel(const bf16* __restrict__ Ow, const float* __restrict__ wo,
                                float* __restrict__ out) {
    const int lane = threadIdx.x;
    const int col = lane & 15, h = lane >> 4;
    const int mrow  = blockIdx.x * 16 + col;
    const int nbase = blockIdx.y * 64;
    const bf16* arow = Ow + (size_t)mrow * HH;

    v8f acc[4];
#pragma unroll
    for (int t = 0; t < 4; ++t)
#pragma unroll
        for (int i = 0; i < 8; ++i) acc[t][i] = 0.0f;

    for (int kb = 0; kb < HH; kb += 32) {
        const v16bf a = frag_a_bf16(arow, kb, h);
#pragma unroll
        for (int t = 0; t < 4; ++t) {
            const v16bf bfr = frag_b_f32(wo + (size_t)(nbase + t * 16 + col) * HH, kb, h);
            acc[t] = WMMA_BF16(a, bfr, acc[t]);
        }
    }
#pragma unroll
    for (int t = 0; t < 4; ++t)
#pragma unroll
        for (int r = 0; r < 8; ++r) {
            const int m = blockIdx.x * 16 + r + 8 * h;
            const int n = nbase + t * 16 + col;
            out[(size_t)m * HH + n] = acc[t][r];
        }
}

// ---------------------------------------------------------------------------
extern "C" void kernel_launch(void* const* d_in, const int* in_sizes, int n_in,
                              void* d_out, int out_size, void* d_ws, size_t ws_size,
                              hipStream_t stream) {
    (void)in_sizes; (void)n_in; (void)out_size; (void)ws_size;
    const float* x    = (const float*)d_in[0];
    const float* y    = (const float*)d_in[1];
    const float* bias = (const float*)d_in[2];
    const float* wq   = (const float*)d_in[3];
    const float* wk   = (const float*)d_in[4];
    const float* wv   = (const float*)d_in[5];
    const float* wo   = (const float*)d_in[6];
    // d_in[7] = training flag (unused, inference path)

    const size_t nElem = (size_t)BB * NHH * LL * DD;   // 4Mi elements each
    bf16* Qw = (bf16*)d_ws;
    bf16* Kw = Qw + nElem;
    bf16* Vw = Kw + nElem;      // transposed [B,NH,D,L]
    bf16* Ow = Vw + nElem;      // merged [B,L,H]

    qkv_proj_kernel<<<dim3((BB * LL) / 16, HH / 64, 3), dim3(32, 1, 1), 0, stream>>>(
        x, y, wq, wk, wv, Qw, Kw, Vw);
    flash_attn_kernel<<<dim3(LL / 64, NHH, BB), dim3(128, 1, 1), 0, stream>>>(
        Qw, Kw, Vw, bias, Ow);
    out_proj_kernel<<<dim3((BB * LL) / 16, HH / 64, 1), dim3(32, 1, 1), 0, stream>>>(
        Ow, wo, (float*)d_out);
}
